// GCNEncoder_49813030699379
// MI455X (gfx1250) — matrix operand order repaired and checked by reading, not compiled, not run on hardware
//
#include <hip/hip_runtime.h>
#include <hip/hip_fp16.h>
#include <stdint.h>

typedef __attribute__((ext_vector_type(16))) _Float16 v16h;
typedef __attribute__((ext_vector_type(8)))  float    v8f;
typedef __attribute__((ext_vector_type(4)))  float    v4f;

#define N_NODES 100000
#define N_EDGES 3200000
#define IN_CH   512
#define HID_CH  40
#define OUT_CH  20
#define M_TILES (N_NODES / 16)   // 6250, exact

// ---------------- degree / normalization ----------------

__global__ void deg_init(float* deg) {
    int i = blockIdx.x * blockDim.x + threadIdx.x;
    if (i < N_NODES) deg[i] = 1.0f;          // self-loop contributes 1
}

__global__ void deg_accum(const int* __restrict__ dst, float* deg) {
    int e = blockIdx.x * blockDim.x + threadIdx.x;
    if (e < N_EDGES) atomicAdd(&deg[dst[e]], 1.0f);
}

__global__ void deg_rsqrt(float* deg) {
    int i = blockIdx.x * blockDim.x + threadIdx.x;
    if (i < N_NODES) deg[i] = rsqrtf(deg[i]);
}

__global__ void edge_norm(const int* __restrict__ src, const int* __restrict__ dst,
                          const float* __restrict__ dinv, float* __restrict__ norm) {
    int e = blockIdx.x * blockDim.x + threadIdx.x;
    if (e < N_EDGES) norm[e] = dinv[src[e]] * dinv[dst[e]];
}

// ---------------- WMMA GEMM: Out[M, NCOLS] = (RELU?relu(A):A)[M, KDIM] @ W[KDIM, NCOLS] --------
// One wave owns a 16-row M-tile and computes all NT 16-col N-tiles (A read once, non-temporal).
// W staged in LDS as f16, layout [n][KPAD] so B fragments are contiguous b128 reads.

template<int KDIM, int KPAD, int NCOLS, int NTILE, bool RELU>
__global__ __launch_bounds__(256) void gemm_wmma(const float* __restrict__ A,
                                                 const float* __restrict__ W,
                                                 float* __restrict__ Out) {
    __shared__ _Float16 ldsW[NTILE * 16 * KPAD];
    const int tid = threadIdx.x;

    // cooperative stage: W[k][n] (f32, row-major [KDIM][NCOLS]) -> ldsW[n*KPAD + k] (f16, padded)
    for (int idx = tid; idx < NTILE * 16 * KPAD; idx += 256) {
        int n = idx / KPAD, k = idx % KPAD;
        float v = 0.0f;
        if (n < NCOLS && k < KDIM) v = W[k * NCOLS + n];
        ldsW[idx] = (_Float16)v;
    }
    __syncthreads();

    const int wave  = tid >> 5;
    const int lane  = tid & 31;
    const int mtile = blockIdx.x * 8 + wave;
    if (mtile >= M_TILES) return;             // wave-uniform: EXEC all-1s inside

    const int half = lane >> 4;
    const int l15  = lane & 15;
    const float* Arow = A + (size_t)(mtile * 16 + l15) * KDIM;

    v8f acc[NTILE] = {};

    for (int kk = 0; kk < KPAD / 32; ++kk) {
        // ---- A fragment: 16x32 f16 (ISA 7.12.2). Per lane two contiguous 8-float K-runs:
        //      run0 = kk*32 + 8*half + [0..7]  -> e[0..7]   (VGPR 0..3)
        //      run1 = kk*32 + 16 + 8*half + [0..7] -> e[8..15] (VGPR 4..7)
        // Loaded as 4 x b128 with NT hint (X / A1 are streamed exactly once). ----
        const int k0 = kk * 32 + 8 * half;
        const int k1 = k0 + 16;
        const v4f z = {0.0f, 0.0f, 0.0f, 0.0f};
        v4f r0 = (k0 + 4  <= KDIM) ? __builtin_nontemporal_load((const v4f*)(Arow + k0))     : z;
        v4f r1 = (k0 + 8  <= KDIM) ? __builtin_nontemporal_load((const v4f*)(Arow + k0 + 4)) : z;
        v4f r2 = (k1 + 4  <= KDIM) ? __builtin_nontemporal_load((const v4f*)(Arow + k1))     : z;
        v4f r3 = (k1 + 8  <= KDIM) ? __builtin_nontemporal_load((const v4f*)(Arow + k1 + 4)) : z;

        float rr[16];
        #pragma unroll
        for (int i = 0; i < 4; ++i) {
            rr[i] = r0[i]; rr[4 + i] = r1[i]; rr[8 + i] = r2[i]; rr[12 + i] = r3[i];
        }
        union { v16h v; _Float16 e[16]; } af;
        #pragma unroll
        for (int i = 0; i < 16; ++i) {
            float x = rr[i];
            if (RELU) x = fmaxf(x, 0.0f);
            af.e[i] = (_Float16)x;
        }

        // ---- NTILE WMMAs reusing the A fragment ----
        #pragma unroll
        for (int t = 0; t < NTILE; ++t) {
            union { v16h v; uint32_t u[8]; } bf;
            const int n = t * 16 + l15;                    // lane -> column
            const uint32_t* bl = reinterpret_cast<const uint32_t*>(
                &ldsW[n * KPAD + kk * 32 + 8 * half]);     // lane-half -> K runs 8*half, 16+8*half
            #pragma unroll
            for (int vv = 0; vv < 4; ++vv) { bf.u[vv] = bl[vv]; bf.u[4 + vv] = bl[8 + vv]; }  // VGPR v<4 -> K=8h+2v,+1; v>=4 -> K=16+8h+2(v-4),+1
            acc[t] = __builtin_amdgcn_wmma_f32_16x16x32_f16(
                false, af.v, false, bf.v, (short)0, acc[t], false, false);
        }
    }

    // ---- store D: lane = N, VGPR g -> M = g + 8*half ----
    #pragma unroll
    for (int t = 0; t < NTILE; ++t) {
        int col = t * 16 + l15;
        if (col < NCOLS) {
            #pragma unroll
            for (int g = 0; g < 8; ++g) {
                int r = mtile * 16 + g + 8 * half;
                Out[(size_t)r * NCOLS + col] = acc[t][g];
            }
        }
    }
}

// ---------------- aggregation ----------------

// Out[node][c] = Hlin[node][c] * dinv[node]^2 + bias[c]   (self-loop message + bias)
template<int CH>
__global__ void self_bias_init(const float* __restrict__ Hlin, const float* __restrict__ dinv,
                               const float* __restrict__ bias, float* __restrict__ Out) {
    int i = blockIdx.x * blockDim.x + threadIdx.x;
    if (i >= N_NODES * CH) return;
    int node = i / CH, c = i % CH;
    float dv = dinv[node];
    Out[i] = Hlin[i] * dv * dv + bias[c];
}

// Out[dst] += Hlin[src] * norm[e]    (one thread per edge; gathers & atomics hit L2)
template<int CH>
__global__ void scatter_add(const int* __restrict__ src, const int* __restrict__ dst,
                            const float* __restrict__ norm, const float* __restrict__ H,
                            float* Out) {
    int e = blockIdx.x * blockDim.x + threadIdx.x;
    if (e >= N_EDGES) return;
    int s = src[e], d = dst[e];
    float w = norm[e];
    const float4* hp = reinterpret_cast<const float4*>(H + (size_t)s * CH);
    float* op = Out + (size_t)d * CH;
    #pragma unroll
    for (int g = 0; g < CH / 4; ++g) {
        float4 hv = hp[g];
        atomicAdd(op + 4 * g + 0, hv.x * w);
        atomicAdd(op + 4 * g + 1, hv.y * w);
        atomicAdd(op + 4 * g + 2, hv.z * w);
        atomicAdd(op + 4 * g + 3, hv.w * w);
    }
}

// ---------------- driver ----------------

extern "C" void kernel_launch(void* const* d_in, const int* in_sizes, int n_in,
                              void* d_out, int out_size, void* d_ws, size_t ws_size,
                              hipStream_t stream) {
    (void)in_sizes; (void)n_in; (void)out_size; (void)ws_size;

    const float* x  = (const float*)d_in[0];
    const int*   ei = (const int*)  d_in[1];
    const float* W1 = (const float*)d_in[2];
    const float* b1 = (const float*)d_in[3];
    const float* W2 = (const float*)d_in[4];
    const float* b2 = (const float*)d_in[5];
    const int* src = ei;
    const int* dst = ei + N_EDGES;

    float* ws   = (float*)d_ws;
    float* dinv = ws;  ws += N_NODES;                    // 0.4 MB
    float* norm = ws;  ws += N_EDGES;                    // 12.8 MB
    float* H1   = ws;  ws += (size_t)N_NODES * HID_CH;   // 16 MB  (x @ W1)
    float* A1   = ws;  ws += (size_t)N_NODES * HID_CH;   // 16 MB  (aggregated conv1, pre-relu)
    float* T2   = ws;  ws += (size_t)N_NODES * OUT_CH;   // 8 MB   (relu(A1) @ W2)
    float* out  = (float*)d_out;

    const int B = 256;
    const int gN  = (N_NODES + B - 1) / B;
    const int gE  = (N_EDGES + B - 1) / B;
    const int gMM = (M_TILES + 7) / 8;                   // 8 waves/block, 1 M-tile/wave

    // normalization (shared by both layers)
    deg_init  <<<gN, B, 0, stream>>>(dinv);
    deg_accum <<<gE, B, 0, stream>>>(dst, dinv);
    deg_rsqrt <<<gN, B, 0, stream>>>(dinv);
    edge_norm <<<gE, B, 0, stream>>>(src, dst, dinv, norm);

    // layer 1: H1 = x @ W1 ; A1 = selfloop+bias ; A1 += scatter(edges)
    gemm_wmma<IN_CH, IN_CH, HID_CH, 3, false><<<gMM, B, 0, stream>>>(x, W1, H1);
    self_bias_init<HID_CH><<<(N_NODES * HID_CH + B - 1) / B, B, 0, stream>>>(H1, dinv, b1, A1);
    scatter_add<HID_CH><<<gE, B, 0, stream>>>(src, dst, norm, H1, A1);

    // layer 2: T2 = relu(A1) @ W2 ; out = selfloop+bias ; out += scatter(edges)
    gemm_wmma<HID_CH, 64, OUT_CH, 2, true><<<gMM, B, 0, stream>>>(A1, W2, T2);
    self_bias_init<OUT_CH><<<(N_NODES * OUT_CH + B - 1) / B, B, 0, stream>>>(T2, dinv, b2, out);
    scatter_add<OUT_CH><<<gE, B, 0, stream>>>(src, dst, norm, T2, out);
}
